// OFS_18262200943202
// MI455X (gfx1250) — matrix operand order repaired and checked by reading, not compiled, not run on hardware
//
#include <hip/hip_runtime.h>

#define C_IN 256
#define NSP  9216   // 96*96 spatial positions
#define MSTEPS (NSP / 32)
#define VROWB 80    // padded LDS row stride in bytes (64B data + 16B TDM pad)

typedef __attribute__((ext_vector_type(16))) __bf16       v16bf;
typedef __attribute__((ext_vector_type(8)))  float        v8f;
typedef __attribute__((ext_vector_type(4)))  unsigned int u32x4;
typedef __attribute__((ext_vector_type(8)))  int          i32x8;
typedef __attribute__((ext_vector_type(4)))  int          i32x4;

union Frag {
  v16bf v;
  u32x4 u[2];
};

static __device__ inline v16bf load_frag(const __bf16* p) {
  Frag f;
  const u32x4* q = (const u32x4*)p;
  f.u[0] = q[0];
  f.u[1] = q[1];
  return f.v;
}

// ---------------------------------------------------------------------------
// Kernel 1: q/k/v projections (1x1 convs). Outputs bf16 in WMMA-friendly
// layouts:
//   qB  : (N, 32)  natural c order      -> B-fragment rows for energy WMMA
//   ksA : (N, 32)  natural c order   -> A-fragment rows for energy WMMA
//   vsB : (C, N)   natural (c-major)    -> TDM-tiled into LDS for output WMMA
// ---------------------------------------------------------------------------
__global__ __launch_bounds__(256) void qkv_kernel(
    const float* __restrict__ x,
    const float* __restrict__ Wq, const float* __restrict__ bq,
    const float* __restrict__ Wk, const float* __restrict__ bk,
    const float* __restrict__ Wv, const float* __restrict__ bv,
    __bf16* __restrict__ qB, __bf16* __restrict__ ksA, __bf16* __restrict__ vsB)
{
  __shared__ float xs[C_IN][32];
  const int n0 = blockIdx.x * 32;
  for (int i = threadIdx.x; i < C_IN * 32; i += 256) {
    int c = i >> 5, n = i & 31;
    xs[c][n] = x[(size_t)c * NSP + n0 + n];
  }
  __syncthreads();

  const int n = threadIdx.x & 31;   // spatial column within tile
  const int g = threadIdx.x >> 5;   // output-channel group 0..7

  for (int j = 0; j < 4; ++j) {     // q projection
    int co = g + 8 * j;
    float acc = bq[co];
    for (int c = 0; c < C_IN; ++c) acc = fmaf(Wq[co * C_IN + c], xs[c][n], acc);
    qB[(size_t)(n0 + n) * 32 + co] = (__bf16)acc;
  }
  // k projection, natural c order for the energy A-fragment
  for (int j = 0; j < 4; ++j) {
    int co = g + 8 * j;
    float acc = bk[co];
    for (int c = 0; c < C_IN; ++c) acc = fmaf(Wk[co * C_IN + c], xs[c][n], acc);
    int pos = co;
    ksA[(size_t)(n0 + n) * 32 + pos] = (__bf16)acc;
  }
  for (int j = 0; j < 32; ++j) {    // v projection, c-major
    int co = g + 8 * j;
    float acc = bv[co];
    for (int c = 0; c < C_IN; ++c) acc = fmaf(Wv[co * C_IN + c], xs[c][n], acc);
    vsB[(size_t)co * NSP + n0 + n] = (__bf16)acc;
  }
}

// ---------------------------------------------------------------------------
// TDM: load one 32(m) x 256(c) bf16 v-tile into LDS, rows padded 64B->80B
// (pad_interval = 16 dwords, pad_amount = 4 dwords) so that the later
// ds_load_b128 fragment reads are bank-conflict-free and 16B-aligned.
// Group1 dword0: data_size=2B (1<<16) | pad_enable (1<<20) |
//                pad_interval=3 (3<<22) | pad_amount=3 (3<<25)
// Toolchain uses the 6-arg builtin: (g0, g1, g2, g3, g4, cpol); groups 2/3/4
// are only used for >2D tensors / gather mode -> zero-filled here.
// ---------------------------------------------------------------------------
static __device__ inline void tdm_load_vtile(const __bf16* gsrc, unsigned ldsoff,
                                             int m0) {
  unsigned long long ga = (unsigned long long)(const void*)gsrc;
  u32x4 g0;
  g0[0] = 1u;                                   // count=1 valid descriptor
  g0[1] = ldsoff;                               // LDS destination offset
  g0[2] = (unsigned)ga;                         // global addr lo
  g0[3] = (unsigned)(ga >> 32) | 0x80000000u;   // global addr hi | type=2
  unsigned td0 = (unsigned)(NSP - m0);          // remaining elems in dim0
  i32x8 g1;
  g1[0] = 0x06D10000;                           // size/pad flags (see above)
  g1[1] = (int)((td0 & 0xFFFFu) << 16);         // tensor_dim0 lo
  g1[2] = (int)((td0 >> 16) | (256u << 16));    // tensor_dim0 hi | tensor_dim1
  g1[3] = (int)(32u << 16);                     // tile_dim0 = 32 (m)
  g1[4] = 256;                                  // tile_dim1 = 256 (c)
  g1[5] = NSP;                                  // tensor_dim0_stride
  g1[6] = 0;
  g1[7] = 0;
  i32x4 gz4 = {0, 0, 0, 0};
  i32x8 gz8 = {0, 0, 0, 0, 0, 0, 0, 0};
  __builtin_amdgcn_tensor_load_to_lds(g0, g1, gz4, gz4, gz8, 0);
}

// ---------------------------------------------------------------------------
// Kernel 2: fused flash-style attention. 4 waves/WG, one 16-row n-block each;
// the waves march over m in lockstep sharing TDM-staged v tiles in LDS.
//   Pass 1: stream k, row max of energy (E^T tiles via WMMA).
//   Pass 2: recompute E^T, P = exp(E - max) packed straight into the
//           A-fragment (layouts coincide), out(16n x 256c) = P x v^T with
//           16 WMMA accumulator tiles fed by ds_load_b128 from LDS.
//   Epilogue: out_co[n] = max_c + mean_c of (out / rowsum).
// ---------------------------------------------------------------------------
__global__ __launch_bounds__(128) void attn_kernel(
    const __bf16* __restrict__ qB, const __bf16* __restrict__ ksA,
    const __bf16* __restrict__ vsB, float* __restrict__ outco)
{
  __shared__ __align__(128) __bf16 vtile[2][C_IN * VROWB / 2];  // 2 x 20KB

  const int lane = threadIdx.x & 31;
  const int wv   = threadIdx.x >> 5;
  const int nb   = blockIdx.x * 4 + wv;     // n-block index (0..575)
  const int n0   = nb * 16;
  const int l15  = lane & 15;
  const int half = lane >> 4;

  const unsigned lds0 = (unsigned)(size_t)(&vtile[0][0]);
  const unsigned lds1 = (unsigned)(size_t)(&vtile[1][0]);

  // q B-fragment for this n-block (loop invariant)
  const v16bf qf = load_frag(qB + (size_t)(n0 + l15) * 32 + half * 16);
  const v8f zero = {};

  // ---- pass 1: row max over all m (k streamed from L2) ----
  float rmax = -3.0e38f;
  for (int m0 = 0; m0 < NSP; m0 += 16) {
    v16bf kf = load_frag(ksA + (size_t)(m0 + l15) * 32 + half * 16);
    v8f e = __builtin_amdgcn_wmma_f32_16x16x32_bf16(
        false, kf, false, qf, (short)0, zero, false, false);
#pragma unroll
    for (int r = 0; r < 8; ++r) rmax = fmaxf(rmax, e[r]);
  }
  rmax = fmaxf(rmax, __shfl_xor(rmax, 16));   // merge m halves; indexed by n=l15

  // ---- pass 2: softmax numerators + output accumulation ----
  if (wv == 0) tdm_load_vtile(vsB, lds0, 0);  // prologue: tile 0 in flight

  v8f acc[16] = {};          // 16 c-tiles of 16x16 f32 (128 VGPRs)
  float rsum = 0.0f;
  for (int i = 0; i < MSTEPS; ++i) {
    const int m0 = i * 32;
    if (wv == 0) {
      if (i + 1 < MSTEPS) {
        tdm_load_vtile(vsB + m0 + 32, (i & 1) ? lds0 : lds1, m0 + 32);
        __builtin_amdgcn_s_wait_tensorcnt((short)1);  // tile i has landed
      } else {
        __builtin_amdgcn_s_wait_tensorcnt((short)0);
      }
    }
    __syncthreads();                          // tile i visible to all waves
    const __bf16* vb = (i & 1) ? &vtile[1][0] : &vtile[0][0];

    v16bf kf0 = load_frag(ksA + (size_t)(m0 + l15) * 32 + half * 16);
    v16bf kf1 = load_frag(ksA + (size_t)(m0 + 16 + l15) * 32 + half * 16);
    v8f e0 = __builtin_amdgcn_wmma_f32_16x16x32_bf16(
        false, kf0, false, qf, (short)0, zero, false, false);
    v8f e1 = __builtin_amdgcn_wmma_f32_16x16x32_bf16(
        false, kf1, false, qf, (short)0, zero, false, false);

    // D-layout of E^T == A-fragment layout of P: pack exp() straight in.
    v16bf aP;
#pragma unroll
    for (int r = 0; r < 8; ++r) {
      float p0 = __expf(e0[r] - rmax);
      float p1 = __expf(e1[r] - rmax);
      rsum += p0 + p1;
      aP[r]     = (__bf16)p0;   // K = m0 + 0..15  (per lane half)
      aP[8 + r] = (__bf16)p1;   // K = m0 + 16..31 (per lane half)
    }

#pragma unroll
    for (int t = 0; t < 16; ++t) {
      // row c = t*16+l15, element offsets within padded row: half*8 and 16+half*8
      const __bf16* lp = vb + (t * 16 + l15) * (VROWB / 2) + half * 8;
      Frag vfr;
      vfr.u[0] = *(const u32x4*)lp;
      vfr.u[1] = *(const u32x4*)(lp + 16);
      v16bf vf = vfr.v;               // 2x ds_load_b128, no conflicts
      acc[t] = __builtin_amdgcn_wmma_f32_16x16x32_bf16(
          false, aP, false, vf, (short)0, acc[t], false, false);
    }
    __syncthreads();                          // all done reading tile i
  }
  rsum += __shfl_xor(rsum, 16);
  float rinv = 1.0f / rsum;    // indexed by n = l15 (both halves hold copies)

  // ---- epilogue: out_co[n] = max_c(out) + mean_c(out) ----
#pragma unroll
  for (int r = 0; r < 8; ++r) {
    float ri = __shfl(rinv, r + 8 * half);   // rowsum for row n = r + 8*half
    float vmax = -3.0e38f, vsum = 0.0f;
#pragma unroll
    for (int t = 0; t < 16; ++t) {
      float v = acc[t][r] * ri;              // this lane's channel c = t*16+l15
      vmax = fmaxf(vmax, v);
      vsum += v;
    }
#pragma unroll
    for (int o = 1; o < 16; o <<= 1) {       // reduce over the 16 channels
      vmax = fmaxf(vmax, __shfl_xor(vmax, o));
      vsum += __shfl_xor(vsum, o);
    }
    if (l15 == 0) outco[n0 + r + 8 * half] = vmax + vsum * (1.0f / 256.0f);
  }
}

// ---------------------------------------------------------------------------
// Kernel 3: spatial gate = softmax(outco * C^-0.5) over N (single block).
// ---------------------------------------------------------------------------
__global__ __launch_bounds__(1024) void gate_kernel(
    const float* __restrict__ outco, float* __restrict__ gate)
{
  __shared__ float sm[32];
  __shared__ float stats[2];
  const int tid = threadIdx.x;
  const float scale = 0.0625f;  // 256^-0.5

  float m = -3.0e38f;
  for (int i = tid; i < NSP; i += 1024) m = fmaxf(m, outco[i] * scale);
#pragma unroll
  for (int o = 16; o >= 1; o >>= 1) m = fmaxf(m, __shfl_xor(m, o));
  if ((tid & 31) == 0) sm[tid >> 5] = m;
  __syncthreads();
  if (tid < 32) {
    float v = sm[tid];
#pragma unroll
    for (int o = 16; o >= 1; o >>= 1) v = fmaxf(v, __shfl_xor(v, o));
    if (tid == 0) stats[0] = v;
  }
  __syncthreads();
  const float M = stats[0];

  float s = 0.0f;
  for (int i = tid; i < NSP; i += 1024) s += __expf(outco[i] * scale - M);
#pragma unroll
  for (int o = 16; o >= 1; o >>= 1) s += __shfl_xor(s, o);
  if ((tid & 31) == 0) sm[tid >> 5] = s;
  __syncthreads();
  if (tid < 32) {
    float v = sm[tid];
#pragma unroll
    for (int o = 16; o >= 1; o >>= 1) v += __shfl_xor(v, o);
    if (tid == 0) stats[1] = v;
  }
  __syncthreads();
  const float inv = 1.0f / stats[1];
  for (int i = tid; i < NSP; i += 1024)
    gate[i] = __expf(outco[i] * scale - M) * inv;
}

// ---------------------------------------------------------------------------
// Kernel 4: y = W6 @ (x * gate) + b6 ; out = (1+gamma)*y + x
// gate is uniform per spatial position -> hoisted out of the dot product.
// ---------------------------------------------------------------------------
__global__ __launch_bounds__(256) void out_kernel(
    const float* __restrict__ x, const float* __restrict__ W6,
    const float* __restrict__ b6, const float* __restrict__ gamma,
    const float* __restrict__ gate, float* __restrict__ out)
{
  __shared__ float xs[C_IN][32];
  const int n0 = blockIdx.x * 32;
  for (int i = threadIdx.x; i < C_IN * 32; i += 256) {
    int c = i >> 5, n = i & 31;
    xs[c][n] = x[(size_t)c * NSP + n0 + n];
  }
  __syncthreads();

  const int n = threadIdx.x & 31;
  const int g = threadIdx.x >> 5;
  const float gt = gate[n0 + n];
  const float gf = 1.0f + gamma[0];

  for (int j = 0; j < 32; ++j) {
    int co = g + 8 * j;
    float acc = 0.0f;
    for (int c = 0; c < C_IN; ++c) acc = fmaf(W6[co * C_IN + c], xs[c][n], acc);
    float y = acc * gt + b6[co];
    out[(size_t)co * NSP + n0 + n] = gf * y + xs[co][n];
  }
}

// ---------------------------------------------------------------------------
extern "C" void kernel_launch(void* const* d_in, const int* in_sizes, int n_in,
                              void* d_out, int out_size, void* d_ws, size_t ws_size,
                              hipStream_t stream)
{
  const float* x     = (const float*)d_in[0];
  const float* Wq    = (const float*)d_in[1];
  const float* bq    = (const float*)d_in[2];
  const float* Wk    = (const float*)d_in[3];
  const float* bk    = (const float*)d_in[4];
  const float* Wv    = (const float*)d_in[5];
  const float* bv    = (const float*)d_in[6];
  const float* W6    = (const float*)d_in[7];
  const float* b6    = (const float*)d_in[8];
  const float* gamma = (const float*)d_in[9];
  float* out = (float*)d_out;

  char* w = (char*)d_ws;
  __bf16* qB  = (__bf16*)w;  w += (size_t)NSP * 32 * 2;     // q^T (N,32) bf16
  __bf16* ksA = (__bf16*)w;  w += (size_t)NSP * 32 * 2;     // k^T (N,32) bf16, natural c order
  __bf16* vsB = (__bf16*)w;  w += (size_t)C_IN * NSP * 2;   // v (C,N) bf16
  float* outco = (float*)w;  w += (size_t)NSP * 4;
  float* gate  = (float*)w;

  qkv_kernel<<<NSP / 32, 256, 0, stream>>>(x, Wq, bq, Wk, bk, Wv, bv,
                                           qB, ksA, vsB);
  attn_kernel<<<NSP / 64, 128, 0, stream>>>(qB, ksA, vsB, outco);
  gate_kernel<<<1, 1024, 0, stream>>>(outco, gate);
  out_kernel<<<NSP / 32, 256, 0, stream>>>(x, W6, b6, gamma, gate, out);
}
